// Crf_18846316494894
// MI455X (gfx1250) — matrix operand order repaired and checked by reading, not compiled, not run on hardware
//
#include <hip/hip_runtime.h>
#include <hip/hip_bf16.h>
#include <math.h>

#define B_  256
#define T_  1024
#define N_  128
#define NT_ 130
#define ST_ 128
#define ED_ 129
#define BB  16     // batches per workgroup in the forward kernel

#define LOG2E 1.44269504088896340736f
#define LN2   0.69314718055994530942f

typedef __attribute__((ext_vector_type(16))) _Float16 v16h;
typedef __attribute__((ext_vector_type(8)))  _Float16 v8h;
typedef __attribute__((ext_vector_type(8)))  float    v8f;

__device__ __forceinline__ v16h cat16(v8h lo, v8h hi) {
  return __builtin_shufflevector(lo, hi, 0,1,2,3,4,5,6,7,8,9,10,11,12,13,14,15);
}

// ---------------------------------------------------------------------------
// Kernel 1: per-batch gold-path score sc[b] and sequence length le[b].
// ---------------------------------------------------------------------------
__global__ __launch_bounds__(256)
void crf_score_kernel(const float* __restrict__ em, const int* __restrict__ tg,
                      const int* __restrict__ mk, const float* __restrict__ tr,
                      float* __restrict__ sc, int* __restrict__ le) {
  __shared__ float sred[256];
  __shared__ int   ired[256];
  const int b = blockIdx.x, tid = threadIdx.x;
  float ps = 0.f; int pl = 0;
  for (int t = tid; t < T_; t += 256) {
    int tag  = tg[b * T_ + t];
    int prev = (t == 0) ? ST_ : tg[b * T_ + t - 1];
    int m    = mk[b * T_ + t];
    float v  = tr[prev * NT_ + tag] + em[((size_t)b * T_ + t) * N_ + tag];
    ps += m ? v : 0.f;
    pl += (m != 0);
  }
  sred[tid] = ps; ired[tid] = pl;
  __syncthreads();
  for (int s = 128; s > 0; s >>= 1) {
    if (tid < s) { sred[tid] += sred[tid + s]; ired[tid] += ired[tid + s]; }
    __syncthreads();
  }
  if (tid == 0) {
    int L  = ired[0];
    int ls = tg[b * T_ + L - 1];
    sc[b] = sred[0] + tr[ls * NT_ + ED_];
    le[b] = L;
  }
}

// ---------------------------------------------------------------------------
// Kernel 2: forward algorithm via WMMA.
//   Per step: S(16x16/wave) = P(16x128) @ E(128x16 tile), P = exp(al - max),
//   al' = max + log(S) + em[:,t,:].  Mask (t < le[b]) is monotone, so the
//   time loop is split at the block-uniform min/max length: the main region
//   is select-free with batched logs; the tail keeps the select; steps past
//   max(le) are identity and skipped entirely.
//   E = exp(tr[:128,:128]) constant in VGPRs, f16 hi+lo compensated:
//   S ~= Ph*Eh + Ph*El + Pl*Eh (f32 accumulate). Two barriers per step.
// ---------------------------------------------------------------------------
__global__ __launch_bounds__(256)
void crf_forward_kernel(const float* __restrict__ em, const float* __restrict__ tr,
                        const float* __restrict__ sc, const int* __restrict__ le,
                        float* __restrict__ zs) {
  __shared__ alignas(16) float    alS[BB][132];     // padded for bank rotation
  __shared__ alignas(16) _Float16 phS[BB][136];
  __shared__ alignas(16) _Float16 plS[BB][136];
  __shared__ alignas(16) float    wmaxS[8][16];     // per-wave row maxima
  __shared__ alignas(16) float    mS[BB];           // per-batch global max

  const int tid  = threadIdx.x;
  const int wave = tid >> 5, lane = tid & 31;
  const int hh   = lane >> 4, ln = lane & 15;
  const int j    = wave * 16 + ln;                  // this lane's tag column
  const int bbase = blockIdx.x * BB;
  const int batch16 = tid >> 4, q = tid & 15;       // cooperative-phase map

  // ---- one-time: E = exp(A) for this wave's 16 columns, f16 hi/lo, B-layout
  v16h Eh[4], El[4];
  #pragma unroll
  for (int c = 0; c < 4; ++c) {
    const int KbB = c * 32 + (hh ? 16 : 0);
    v16h eh, el;
    #pragma unroll
    for (int e = 0; e < 16; ++e) {
      float x = expf(tr[(KbB + e) * NT_ + j]);
      _Float16 xh = (_Float16)x;
      eh[e] = xh;
      el[e] = (_Float16)(x - (float)xh);
    }
    Eh[c] = eh; El[c] = el;
  }

  // sequence lengths: per-lane rows + block-uniform bounds (uniform -> SGPR)
  int lev[8];
  #pragma unroll
  for (int r = 0; r < 8; ++r) lev[r] = le[bbase + r + 8 * hh];
  int tcut = T_, tmax = 0;
  #pragma unroll
  for (int i = 0; i < BB; ++i) {
    int L = le[bbase + i];
    tcut = (L < tcut) ? L : tcut;
    tmax = (L > tmax) ? L : tmax;
  }

  // ---- t = 0 init: al = tr[ST, j] + em[b, 0, j]
  float alv[8];
  const float a0j = tr[ST_ * NT_ + j];
  #pragma unroll
  for (int r = 0; r < 8; ++r) {
    int bg = bbase + r + 8 * hh;
    alv[r] = a0j + em[((size_t)bg * T_) * N_ + j];
  }

  // One forward step. MASKED==0: unconditional update, batched logs.
#define CRF_STEP(MASKED)                                                       \
  {                                                                            \
    float ev[8];                                                               \
    _Pragma("unroll")                                                          \
    for (int r = 0; r < 8; ++r) {                                              \
      int bg = bbase + r + 8 * hh;                                             \
      ev[r] = em[((size_t)bg * T_ + t) * N_ + j];                              \
    }                                                                          \
    _Pragma("unroll")                                                          \
    for (int r = 0; r < 8; ++r) alS[r + 8 * hh][j] = alv[r];                   \
    _Pragma("unroll")                                                          \
    for (int r = 0; r < 8; ++r) {                                              \
      float mx = alv[r];                                                       \
      _Pragma("unroll")                                                        \
      for (int s = 8; s > 0; s >>= 1) mx = fmaxf(mx, __shfl_xor(mx, s, 32));   \
      if (ln == 0) wmaxS[wave][r + 8 * hh] = mx;                               \
    }                                                                          \
    __syncthreads();                                                           \
    {                                                                          \
      float m = wmaxS[0][batch16];                                             \
      _Pragma("unroll")                                                        \
      for (int w = 1; w < 8; ++w) m = fmaxf(m, wmaxS[w][batch16]);             \
      if (q == 0) mS[batch16] = m;                                             \
      _Pragma("unroll")                                                        \
      for (int k = 0; k < 4; ++k) {                                            \
        int j0 = 2 * q + 32 * k;                                               \
        float p0 = __builtin_amdgcn_exp2f((alS[batch16][j0]     - m) * LOG2E); \
        float p1 = __builtin_amdgcn_exp2f((alS[batch16][j0 + 1] - m) * LOG2E); \
        _Float16 h0 = (_Float16)p0, h1 = (_Float16)p1;                         \
        phS[batch16][j0]     = h0;                                             \
        phS[batch16][j0 + 1] = h1;                                             \
        plS[batch16][j0]     = (_Float16)(p0 - (float)h0);                     \
        plS[batch16][j0 + 1] = (_Float16)(p1 - (float)h1);                     \
      }                                                                        \
    }                                                                          \
    __syncthreads();                                                           \
    v16h ah[4], av[4];                                                         \
    _Pragma("unroll")                                                          \
    for (int c = 0; c < 4; ++c) {                                              \
      const int KbA = c * 32 + (hh ? 16 : 0);                                  \
      ah[c] = cat16(*(const v8h*)&phS[ln][KbA],                                \
                    *(const v8h*)&phS[ln][KbA + 8]);                           \
      av[c] = cat16(*(const v8h*)&plS[ln][KbA],                                \
                    *(const v8h*)&plS[ln][KbA + 8]);                           \
    }                                                                          \
    v8f acc = {};                                                              \
    _Pragma("unroll")                                                          \
    for (int c = 0; c < 4; ++c) {                                              \
      acc = __builtin_amdgcn_wmma_f32_16x16x32_f16(false, ah[c], false, Eh[c], \
                                                   (short)0, acc, false, false);\
      acc = __builtin_amdgcn_wmma_f32_16x16x32_f16(false, ah[c], false, El[c], \
                                                   (short)0, acc, false, false);\
      acc = __builtin_amdgcn_wmma_f32_16x16x32_f16(false, av[c], false, Eh[c], \
                                                   (short)0, acc, false, false);\
    }                                                                          \
    const float4 m0 = *(const float4*)&mS[8 * hh];                             \
    const float4 m1 = *(const float4*)&mS[8 * hh + 4];                         \
    const float mrow[8] = {m0.x, m0.y, m0.z, m0.w, m1.x, m1.y, m1.z, m1.w};    \
    float lg[8];                                                               \
    _Pragma("unroll")                                                          \
    for (int r = 0; r < 8; ++r) lg[r] = __builtin_amdgcn_logf(acc[r]);         \
    _Pragma("unroll")                                                          \
    for (int r = 0; r < 8; ++r) {                                              \
      float cand = mrow[r] + lg[r] * LN2 + ev[r];                              \
      if (MASKED) alv[r] = (t < lev[r]) ? cand : alv[r];                       \
      else        alv[r] = cand;                                               \
    }                                                                          \
  }

  for (int t = 1; t < tcut; ++t) CRF_STEP(0)   // all batches active
  for (int t = tcut; t < tmax; ++t) CRF_STEP(1) // tail with per-batch mask
#undef CRF_STEP

  // ---- finalize: z[b] = logsumexp_j(al[b,j] + tr[j, ED]); emit z - sc
  const float trE = tr[j * NT_ + ED_];
  #pragma unroll
  for (int r = 0; r < 8; ++r) alS[r + 8 * hh][j] = alv[r] + trE;
  __syncthreads();
  float mx = -3.0e38f;
  #pragma unroll
  for (int k = 0; k < 8; ++k) mx = fmaxf(mx, alS[batch16][q + 16 * k]);
  #pragma unroll
  for (int s = 8; s > 0; s >>= 1) mx = fmaxf(mx, __shfl_xor(mx, s, 32));
  float sm = 0.f;
  #pragma unroll
  for (int k = 0; k < 8; ++k)
    sm += __builtin_amdgcn_exp2f((alS[batch16][q + 16 * k] - mx) * LOG2E);
  #pragma unroll
  for (int s = 8; s > 0; s >>= 1) sm += __shfl_xor(sm, s, 32);
  if (q == 0) {
    int bg = bbase + batch16;
    zs[bg] = mx + __builtin_amdgcn_logf(sm) * LN2 - sc[bg];
  }
}

// ---------------------------------------------------------------------------
// Kernel 3: deterministic mean over the 256 per-batch (z - sc) values.
// ---------------------------------------------------------------------------
__global__ __launch_bounds__(256)
void crf_mean_kernel(const float* __restrict__ zs, float* __restrict__ out) {
  __shared__ float sred[256];
  const int tid = threadIdx.x;
  sred[tid] = zs[tid];
  __syncthreads();
  for (int s = 128; s > 0; s >>= 1) {
    if (tid < s) sred[tid] += sred[tid + s];
    __syncthreads();
  }
  if (tid == 0) out[0] = sred[0] * (1.0f / (float)B_);
}

// ---------------------------------------------------------------------------
extern "C" void kernel_launch(void* const* d_in, const int* in_sizes, int n_in,
                              void* d_out, int out_size, void* d_ws, size_t ws_size,
                              hipStream_t stream) {
  (void)in_sizes; (void)n_in; (void)out_size; (void)ws_size;
  const float* em = (const float*)d_in[0];   // (B,T,N) f32
  const int*   tg = (const int*)d_in[1];     // (B,T)   i32
  const int*   mk = (const int*)d_in[2];     // (B,T)   bool -> int
  const float* tr = (const float*)d_in[3];   // (N+2,N+2) f32

  float* ws = (float*)d_ws;
  float* sc = ws;                    // B floats
  int*   le = (int*)(ws + B_);       // B ints
  float* zs = ws + 2 * B_;           // B floats

  crf_score_kernel<<<B_, 256, 0, stream>>>(em, tg, mk, tr, sc, le);
  crf_forward_kernel<<<B_ / BB, 256, 0, stream>>>(em, tr, sc, le, zs);
  crf_mean_kernel<<<1, 256, 0, stream>>>(zs, (float*)d_out);
}
